// TransformerEncoder_82987358093748
// MI455X (gfx1250) — matrix operand order repaired and checked by reading, not compiled, not run on hardware
//
#include <hip/hip_runtime.h>
#include <math.h>
#include <stdint.h>

// ---------------------------------------------------------------------------
// CDNA5 (gfx1250) transformer encoder forward.
// GEMMs: TDM (tensor_load_to_lds) staging + v_wmma_f32_16x16x32_bf16.
// ---------------------------------------------------------------------------

typedef __attribute__((ext_vector_type(16))) __bf16 v16bf;
typedef __attribute__((ext_vector_type(8)))  float  v8f;
typedef __attribute__((ext_vector_type(4)))  float  v4f;
typedef __attribute__((ext_vector_type(4)))  unsigned int v4u;
typedef __attribute__((ext_vector_type(8)))  int    v8i;
typedef __attribute__((ext_vector_type(4)))  int    v4i;

// float -> bf16 round-to-nearest-even via integer ops
__device__ __forceinline__ __bf16 f2bf(float f) {
  unsigned int u = __builtin_bit_cast(unsigned int, f);
  u += 0x7FFFu + ((u >> 16) & 1u);
  unsigned short s = (unsigned short)(u >> 16);
  return __builtin_bit_cast(__bf16, s);
}

// ---------------------------------------------------------------------------
// TDM: DMA a [rows x 32] f32 tile (row stride = stride_elems f32) from global
// into LDS at lds_byte_addr, padding each 32-DWORD row with 4 DWORDs so the
// LDS row stride is 36 floats (144 B, 16B-aligned rows for v4f fragment loads).
// D# per CDNA5 ISA 8.3-8.5; groups 2/3 zero (tile_dim3/4 = 0 -> unused).
// 6-arg builtin form (this toolchain): (v4u, v8i, v4i, v4i, v8i, i32 cpol).
// ---------------------------------------------------------------------------
__device__ __forceinline__ void tdm_load_tile32_f32(
    unsigned lds_byte_addr, const void* gptr, int rows, int stride_elems)
{
  unsigned long long ga = (unsigned long long)(uintptr_t)gptr;
  v4u g0;
  g0.x = 1u;                                                 // count=1 (valid)
  g0.y = lds_byte_addr;                                      // lds_addr
  g0.z = (unsigned)ga;                                       // global_addr[31:0]
  g0.w = (unsigned)((ga >> 32) & 0x01FFFFFFu) | (2u << 30);  // addr[56:32]|type=2
  v8i g1;
  // data_size=2(4B) | pad_enable | pad_interval=4 (32 DW) | pad_amount=3 (4 DW)
  g1[0] = (int)((2u << 16) | (1u << 20) | (4u << 22) | (3u << 25));
  g1[1] = (int)(32u << 16);                                  // tensor_dim0=32 (lo16)
  g1[2] = (int)(((unsigned)rows & 0xFFFFu) << 16);           // dim0 hi | tensor_dim1 lo
  g1[3] = (int)((((unsigned)rows >> 16) & 0xFFFFu) | (32u << 16)); // dim1 hi | tile_dim0=32
  g1[4] = (int)((unsigned)rows & 0xFFFFu);                   // tile_dim1 | tile_dim2=0
  g1[5] = (int)(unsigned)stride_elems;                       // tensor_dim0_stride lo32
  g1[6] = 0;                                                 // stride hi | dim1_stride lo
  g1[7] = 0;
  v4i z4 = {0, 0, 0, 0};
  v8i z8 = {0, 0, 0, 0, 0, 0, 0, 0};
  __builtin_amdgcn_tensor_load_to_lds(g0, g1, z4, z4, z8, 0);
}

// ---------------------------------------------------------------------------
// GEMM: C[M,N] = act(A[M,K] @ W[N,K]^T + bias[N] (+ bias2[N]))
// Block: 256 threads = 8 waves; tile 64(M) x 64(N), K step 32.
// Wave 0 DMAs both tiles via TDM, waits TENSORcnt, barrier publishes LDS.
// ACT: 0=none, 1=exact GELU, 2=ReLU
// ---------------------------------------------------------------------------
template<int ACT, bool BIAS2>
__global__ __launch_bounds__(256)
void gemm_bt_kernel(const float* __restrict__ A, const float* __restrict__ W,
                    const float* __restrict__ bias, const float* __restrict__ bias2,
                    float* __restrict__ C, int M, int N, int K)
{
  __shared__ float Asf[64][36];   // 32 data + 4 TDM pad floats per row
  __shared__ float Bsf[64][36];

  const int t    = threadIdx.x;
  const int lane = t & 31;
  const int wid  = t >> 5;
  const int wm   = wid & 3;       // M subtile 0..3
  const int wn   = wid >> 2;      // N group   0..1 (32 cols)
  const int hf   = lane >> 4;     // lane half
  const int r    = lane & 15;
  const int m0   = blockIdx.y * 64;
  const int n0   = blockIdx.x * 64;

  const unsigned asf_base = (unsigned)(uintptr_t)&Asf[0][0];
  const unsigned bsf_base = (unsigned)(uintptr_t)&Bsf[0][0];

  v8f acc[2] = {v8f{}, v8f{}};

  for (int k0 = 0; k0 < K; k0 += 32) {
    __syncthreads();              // previous tiles fully consumed
    if (wid == 0) {               // wave-uniform branch; TDM ignores EXEC
      tdm_load_tile32_f32(asf_base, A + (size_t)m0 * K + k0, 64, K);
      tdm_load_tile32_f32(bsf_base, W + (size_t)n0 * K + k0, 64, K);
      __builtin_amdgcn_s_wait_tensorcnt(0);
    }
    __syncthreads();              // tiles visible to all 8 waves

    // A fragment: 16x32 (ISA 7.12.2), f32 LDS -> bf16
    v16bf afr;
    {
      const float* arow = &Asf[wm * 16 + r][0];
      v4f a0 = *(const v4f*)(arow + 8 * hf);
      v4f a1 = *(const v4f*)(arow + 8 * hf + 4);
      v4f a2 = *(const v4f*)(arow + 16 + 8 * hf);
      v4f a3 = *(const v4f*)(arow + 16 + 8 * hf + 4);
      #pragma unroll
      for (int j = 0; j < 4; ++j) {
        afr[j]      = f2bf(a0[j]);
        afr[j + 4]  = f2bf(a1[j]);
        afr[j + 8]  = f2bf(a2[j]);
        afr[j + 12] = f2bf(a3[j]);
      }
    }
    #pragma unroll
    for (int nt = 0; nt < 2; ++nt) {
      // B fragment: 32x16, K = 8*hf + j (j<8), 16 + 8*hf + (j-8) (j>=8), N = r
      const float* brow = &Bsf[wn * 32 + nt * 16 + r][0];
      v4f b0 = *(const v4f*)(brow + 8 * hf);
      v4f b1 = *(const v4f*)(brow + 8 * hf + 4);
      v4f b2 = *(const v4f*)(brow + 16 + 8 * hf);
      v4f b3 = *(const v4f*)(brow + 16 + 8 * hf + 4);
      v16bf bfr;
      #pragma unroll
      for (int j = 0; j < 4; ++j) {
        bfr[j]      = f2bf(b0[j]);
        bfr[j + 4]  = f2bf(b1[j]);
        bfr[j + 8]  = f2bf(b2[j]);
        bfr[j + 12] = f2bf(b3[j]);
      }
      acc[nt] = __builtin_amdgcn_wmma_f32_16x16x32_bf16(
          false, afr, false, bfr, (short)0, acc[nt], false, false);
    }
  }

  // epilogue: bias (+bias2) + activation, scatter C fragment (M = e + 8*hf)
  #pragma unroll
  for (int nt = 0; nt < 2; ++nt) {
    const int col = n0 + wn * 32 + nt * 16 + r;
    float bsum = bias[col];
    if (BIAS2) bsum += bias2[col];
    #pragma unroll
    for (int e = 0; e < 8; ++e) {
      const int row = m0 + wm * 16 + e + 8 * hf;
      float val = acc[nt][e] + bsum;
      if (ACT == 1) val = 0.5f * val * (1.0f + erff(val * 0.70710678118654752f));
      else if (ACT == 2) val = fmaxf(val, 0.0f);
      C[(size_t)row * N + col] = val;
    }
  }
  (void)M;
}

// ---------------------------------------------------------------------------
// Attention: grid (S/32, H, B), block 64 = 2 waves; each wave owns 16 q rows.
// scores(16x512) via 32 WMMAs -> LDS f32 -> in-wave softmax -> P@V via 32 WMMAs.
// ---------------------------------------------------------------------------
__global__ __launch_bounds__(64)
void attention_kernel(const float* __restrict__ Q, const float* __restrict__ Km,
                      const float* __restrict__ Vm, float* __restrict__ O, float scale)
{
  const int S = 512, E = 256, Dh = 32;
  __shared__ float sc[32][512];   // 64 KB: two waves x 16 rows x 512 scores

  const int t    = threadIdx.x;
  const int lane = t & 31;
  const int w    = t >> 5;        // wave 0..1
  const int hf   = lane >> 4;
  const int r    = lane & 15;
  const int b = blockIdx.z, h = blockIdx.y, qb = blockIdx.x;

  // Q fragment (A, 16x32): row = q token, K = head dim; fold softmax scale here
  const float* qp = Q + (size_t)(b * S + qb * 32 + w * 16 + r) * E + h * Dh;
  v16bf aq;
  #pragma unroll
  for (int j = 0; j < 8; ++j) {
    aq[j]     = f2bf(qp[8 * hf + j] * scale);
    aq[j + 8] = f2bf(qp[16 + 8 * hf + j] * scale);
  }

  // scores = (scale*q) @ k^T : one WMMA per 16 k-rows (K dim = Dh = 32)
  for (int kt = 0; kt < S / 16; ++kt) {
    const float* kp = Km + (size_t)(b * S + kt * 16 + r) * E + h * Dh;
    v16bf bk;
    #pragma unroll
    for (int j = 0; j < 8; ++j) { bk[j] = f2bf(kp[8 * hf + j]); bk[j + 8] = f2bf(kp[16 + 8 * hf + j]); }
    v8f c = {};
    c = __builtin_amdgcn_wmma_f32_16x16x32_bf16(false, aq, false, bk, (short)0, c, false, false);
    #pragma unroll
    for (int e = 0; e < 8; ++e) sc[w * 16 + e + 8 * hf][kt * 16 + r] = c[e];
  }

  // softmax: lanes (r, r+16) pair on row w*16+r, split columns, combine via shfl_xor
  {
    float* row = &sc[w * 16 + r][0];
    const int c0 = hf * 256;
    float mx = -3.4e38f;
    for (int j = 0; j < 256; ++j) mx = fmaxf(mx, row[c0 + j]);
    mx = fmaxf(mx, __shfl_xor(mx, 16, 32));
    float sum = 0.f;
    for (int j = 0; j < 256; ++j) { float e = __expf(row[c0 + j] - mx); row[c0 + j] = e; sum += e; }
    sum += __shfl_xor(sum, 16, 32);
    const float inv = 1.0f / sum;
    for (int j = 0; j < 256; ++j) row[c0 + j] *= inv;
  }

  // attended = P @ V : K loop over S, two 16-col N tiles (Dh = 32)
  #pragma unroll
  for (int nt = 0; nt < 2; ++nt) {
    v8f acc = {};
    for (int k0 = 0; k0 < S; k0 += 32) {
      const float* prow = &sc[w * 16 + r][k0];
      v16bf ap;
      #pragma unroll
      for (int j = 0; j < 8; ++j) {
        ap[j]     = f2bf(prow[8 * hf + j]);
        ap[j + 8] = f2bf(prow[16 + 8 * hf + j]);
      }
      const float* vp = Vm + (size_t)(b * S + k0) * E + h * Dh + nt * 16 + r;
      v16bf bvf;
      #pragma unroll
      for (int j = 0; j < 8; ++j) { bvf[j] = f2bf(vp[(size_t)(8 * hf + j) * E]); bvf[j + 8] = f2bf(vp[(size_t)(16 + 8 * hf + j) * E]); }
      acc = __builtin_amdgcn_wmma_f32_16x16x32_bf16(false, ap, false, bvf, (short)0, acc, false, false);
    }
    #pragma unroll
    for (int e = 0; e < 8; ++e)
      O[(size_t)(b * S + qb * 32 + w * 16 + e + 8 * hf) * E + h * Dh + nt * 16 + r] = acc[e];
  }
}

// ---------------------------------------------------------------------------
// LayerNorm (+ optional residual), one 256-thread block per row (E = 256)
// ---------------------------------------------------------------------------
__global__ __launch_bounds__(256)
void ln_kernel(const float* __restrict__ xin, const float* __restrict__ res,
               const float* __restrict__ gw, const float* __restrict__ gb,
               float* __restrict__ out)
{
  const int E = 256;
  const int row = blockIdx.x, t = threadIdx.x;
  const size_t base = (size_t)row * E;
  float v = xin[base + t];
  if (res) v += res[base + t];

  __shared__ float red[256];
  red[t] = v; __syncthreads();
  #pragma unroll
  for (int s = 128; s > 0; s >>= 1) { if (t < s) red[t] += red[t + s]; __syncthreads(); }
  const float mu = red[0] * (1.0f / E);
  __syncthreads();
  const float d = v - mu;
  red[t] = d * d; __syncthreads();
  #pragma unroll
  for (int s = 128; s > 0; s >>= 1) { if (t < s) red[t] += red[t + s]; __syncthreads(); }
  const float var = red[0] * (1.0f / E);
  out[base + t] = d * rsqrtf(var + 1e-5f) * gw[t] + gb[t];
}

// ---------------------------------------------------------------------------
// Host orchestration
// ---------------------------------------------------------------------------
extern "C" void kernel_launch(void* const* d_in, const int* in_sizes, int n_in,
                              void* d_out, int out_size, void* d_ws, size_t ws_size,
                              hipStream_t stream)
{
  (void)in_sizes; (void)n_in; (void)out_size; (void)ws_size;
  const float* cnn_features = (const float*)d_in[0];
  const float* cnn_w   = (const float*)d_in[1];
  const float* cnn_b   = (const float*)d_in[2];
  const float* tok_emb = (const float*)d_in[3];
  const float* Wq = (const float*)d_in[4];  const float* bq = (const float*)d_in[5];
  const float* Wk = (const float*)d_in[6];  const float* bk = (const float*)d_in[7];
  const float* Wv = (const float*)d_in[8];  const float* bv = (const float*)d_in[9];
  const float* Wo = (const float*)d_in[10]; const float* bo = (const float*)d_in[11];
  const float* ln1_w = (const float*)d_in[12]; const float* ln1_b = (const float*)d_in[13];
  const float* ln2_w = (const float*)d_in[14]; const float* ln2_b = (const float*)d_in[15];
  const float* mlp_w1 = (const float*)d_in[16]; const float* mlp_b1 = (const float*)d_in[17];
  const float* mlp_w2 = (const float*)d_in[18]; const float* mlp_b2 = (const float*)d_in[19];
  const float* lnf_w = (const float*)d_in[20]; const float* lnf_b = (const float*)d_in[21];
  const float* out_w1 = (const float*)d_in[22]; const float* out_b1 = (const float*)d_in[23];
  const float* out_w2 = (const float*)d_in[24]; const float* out_b2 = (const float*)d_in[25];

  const int Mtok = 32 * 512;        // 16384 tokens
  const int E = 256, L = 4;
  const float scale = 0.17677669529663687f;  // 1/sqrt(32)

  float* ws = (float*)d_ws;
  const size_t seg = (size_t)Mtok * E;
  float* x    = ws;                 // [M,E]
  float* qb_  = ws + 1 * seg;       // [M,E]
  float* kb_  = ws + 2 * seg;       // [M,E]
  float* vb_  = ws + 3 * seg;       // [M,E]
  float* t0   = ws + 4 * seg;       // [M,E]
  float* t1   = ws + 5 * seg;       // [M,E]
  float* hbuf = qb_;                // [M,768] aliases q/k/v (dead when MLP runs)

  const dim3 blk(256);

  // x = cnn_features @ cnn_w^T + cnn_b + tok_emb[0]
  gemm_bt_kernel<0, true><<<dim3(E / 64, Mtok / 64), blk, 0, stream>>>(
      cnn_features, cnn_w, cnn_b, tok_emb, x, Mtok, E, 512);

  for (int i = 0; i < L; ++i) {
    const float* Wq_i = Wq + (size_t)i * E * E;
    const float* Wk_i = Wk + (size_t)i * E * E;
    const float* Wv_i = Wv + (size_t)i * E * E;
    const float* Wo_i = Wo + (size_t)i * E * E;

    gemm_bt_kernel<0, false><<<dim3(E / 64, Mtok / 64), blk, 0, stream>>>(
        x, Wq_i, bq + i * E, nullptr, qb_, Mtok, E, E);
    gemm_bt_kernel<0, false><<<dim3(E / 64, Mtok / 64), blk, 0, stream>>>(
        x, Wk_i, bk + i * E, nullptr, kb_, Mtok, E, E);
    gemm_bt_kernel<0, false><<<dim3(E / 64, Mtok / 64), blk, 0, stream>>>(
        x, Wv_i, bv + i * E, nullptr, vb_, Mtok, E, E);

    attention_kernel<<<dim3(16, 8, 32), dim3(64), 0, stream>>>(qb_, kb_, vb_, t0, scale);

    gemm_bt_kernel<0, false><<<dim3(E / 64, Mtok / 64), blk, 0, stream>>>(
        t0, Wo_i, bo + i * E, nullptr, t1, Mtok, E, E);
    ln_kernel<<<Mtok, 256, 0, stream>>>(x, t1, ln1_w + i * E, ln1_b + i * E, x);

    gemm_bt_kernel<1, false><<<dim3(768 / 64, Mtok / 64), blk, 0, stream>>>(
        x, mlp_w1 + (size_t)i * 768 * E, mlp_b1 + i * 768, nullptr, hbuf, Mtok, 768, E);
    gemm_bt_kernel<0, false><<<dim3(E / 64, Mtok / 64), blk, 0, stream>>>(
        hbuf, mlp_w2 + (size_t)i * E * 768, mlp_b2 + i * E, nullptr, t1, Mtok, E, 768);
    ln_kernel<<<Mtok, 256, 0, stream>>>(x, t1, ln2_w + i * E, ln2_b + i * E, x);
  }

  // final LN + output head
  ln_kernel<<<Mtok, 256, 0, stream>>>(x, nullptr, lnf_w, lnf_b, t0);
  gemm_bt_kernel<1, false><<<dim3(512 / 64, Mtok / 64), blk, 0, stream>>>(
      t0, out_w1, out_b1, nullptr, hbuf, Mtok, 512, E);
  gemm_bt_kernel<2, false><<<dim3(512 / 64, Mtok / 64), blk, 0, stream>>>(
      hbuf, out_w2, out_b2, nullptr, (float*)d_out, Mtok, 512, 512);
}
